// HierarchicalHAGEN_7370163880317
// MI455X (gfx1250) — matrix operand order repaired and checked by reading, not compiled, not run on hardware
//
#include <hip/hip_runtime.h>
#include <hip/hip_bf16.h>

// CDNA5 / gfx1250 wave32 WMMA implementation of hierarchical GCN+LSTM.
// Matrix work mapped onto v_wmma_f32_16x16x32_f16 (f16 in, f32 accumulate).

typedef __attribute__((ext_vector_type(16))) _Float16 v16h;
typedef __attribute__((ext_vector_type(8)))  float    v8f;

#define INCH   16
#define HID    64
#define GATES  256        // 4*HID
#define TSTEPS 52

__device__ __forceinline__ float sigf(float x) { return 1.0f / (1.0f + __expf(-x)); }

// ---- WMMA fragment index helpers (CDNA5 ISA §7.12.2, wave32) ----
// A (16x32 f16, MxK): lane m = l&15; lanes 0-15: V0..3 K=0..7, V4..7 K=16..23;
// lanes 16-31 shift K by +8 within each half.
__device__ __forceinline__ int a_kidx(int lane, int j) {
  int kb = (lane & 16) ? 8 : 0;
  return (j < 4) ? (kb + 2 * j) : (16 + kb + 2 * (j - 4));
}
// B (32x16 f16, KxN): lane n = l&15; same K layout as A: lanes 0-15: V0..3 K=0..7,
// V4..7 K=16..23; lanes 16-31 shift K by +8 within each half.
__device__ __forceinline__ int b_kidx(int lane, int j) {
  int kb = (lane & 16) ? 8 : 0;
  return (j < 4) ? (kb + 2 * j) : (16 + kb + 2 * (j - 4));
}
// C/D (16x16 f32): lane n = l&15; VGPR r -> row m = r + (lane<16 ? 0 : 8).

// ============================ degree / norm =================================
__global__ void deg_init_kernel(float* deg, int N) {
  int i = blockIdx.x * blockDim.x + threadIdx.x;
  if (i < N) deg[i] = 1.0f;                       // self-loop
}
__global__ void deg_accum_kernel(const int* __restrict__ dst, float* deg, int E) {
  int i = blockIdx.x * blockDim.x + threadIdx.x;
  if (i < E) atomicAdd(&deg[dst[i]], 1.0f);
}
__global__ void deg_rsqrt_kernel(float* deg, int N) {
  int i = blockIdx.x * blockDim.x + threadIdx.x;
  if (i < N) deg[i] = rsqrtf(deg[i]);             // deg >= 1 always
}

// ====================== xw = x @ gcn_w  (WMMA, K=16 pad 32) =================
// block = 128 threads = 4 waves; wave w computes cols [16w,16w+16) of 16 rows.
// Branch-free: row index clamped for loads; K>=16 half of A is zero so the
// matching B elements (8-15, K=16..31) are don't-care and loaded unguarded.
__global__ __launch_bounds__(128) void gemm_xw_kernel(
    const float* __restrict__ x,   // [R,16]
    const float* __restrict__ gw,  // [16,64]
    float* __restrict__ xw,        // [R,64]
    int R) {
  int lane = threadIdx.x & 31;
  int wave = threadIdx.x >> 5;                    // col tile 0..3
  int row0 = blockIdx.x * 16;
  int m    = lane & 15;
  int row  = row0 + m;
  if (row >= R) row = R - 1;                      // clamp: loads stay in-bounds
  int kb   = (lane & 16) ? 8 : 0;

  // A: lanes<16 need x[row][0..7], lanes>=16 need x[row][8..15]; K>=16 -> 0.
  const float4* xr = (const float4*)(x + (size_t)row * INCH + kb);
  float4 p0 = xr[0], p1 = xr[1];
  v16h a;
  a[0] = (_Float16)p0.x; a[1] = (_Float16)p0.y;
  a[2] = (_Float16)p0.z; a[3] = (_Float16)p0.w;
  a[4] = (_Float16)p1.x; a[5] = (_Float16)p1.y;
  a[6] = (_Float16)p1.z; a[7] = (_Float16)p1.w;
  #pragma unroll
  for (int j = 8; j < 16; ++j) a[j] = (_Float16)0.f;

  int col = wave * 16 + m;
  v16h b;                                         // K&15 everywhere: upper-lane
  #pragma unroll                                  // values are don't-care.
  for (int j = 0; j < 8; ++j) {
    b[2*j]   = (_Float16)gw[(size_t)((kb + 2*j) & 15)     * HID + col];
    b[2*j+1] = (_Float16)gw[(size_t)((kb + 2*j + 1) & 15) * HID + col];
  }

  v8f acc = {};
  acc = __builtin_amdgcn_wmma_f32_16x16x32_f16(false, a, false, b, (short)0, acc, false, false);

  int mb = (lane & 16) ? 8 : 0;
  float* op = xw + (size_t)(row0 + mb) * HID + col;
  if (row0 + 16 <= R) {                           // uniform fast path
    #pragma unroll
    for (int r = 0; r < 8; ++r) op[(size_t)r * HID] = acc[r];
  } else {                                        // tail tile only
    #pragma unroll
    for (int r = 0; r < 8; ++r)
      if (row0 + mb + r < R) op[(size_t)r * HID] = acc[r];
  }
}

// ============== hseq[t*N+n] = dinv[n]^2 * xw[n*T+t]  (self loop) ============
// grid: (ceil(N*64/256), T) -> no integer division anywhere.
__global__ void hseq_init_kernel(const float* __restrict__ xw,
                                 const float* __restrict__ dinv,
                                 float* __restrict__ hseq, int N) {
  int t = blockIdx.y;
  long long i = blockIdx.x * 256LL + threadIdx.x;  // over N*64
  if (i >= (long long)N * HID) return;
  int n = (int)(i >> 6), h = (int)(i & 63);
  float dv = dinv[n];
  hseq[(size_t)t * N * HID + i] = dv * dv * xw[((size_t)n * TSTEPS + t) * HID + h];
}

// ======== edge scatter: hseq[t*N+d] += dinv[s]*dinv[d]*xw[s*T+t] ============
// grid: (ceil(E/4), T); 64 consecutive threads = 64 channels of one (e,t)
// -> coalesced global_load_b32 + coalesced global_atomic_add_f32.
__global__ __launch_bounds__(256) void agg_edges_kernel(
    const int* __restrict__ src, const int* __restrict__ dst,
    const float* __restrict__ dinv,
    const float* __restrict__ xw, float* __restrict__ hseq,
    int N, int E) {
  int t = blockIdx.y;
  int e = blockIdx.x * 4 + (threadIdx.x >> 6);
  int h = threadIdx.x & 63;
  if (e >= E) return;
  int s = src[e], d = dst[e];
  float w = dinv[s] * dinv[d];
  float v = xw[((size_t)s * TSTEPS + t) * HID + h];
  atomicAdd(&hseq[((size_t)t * N + d) * HID + h], w * v);
}

// ===================== fused LSTM + final FC (WMMA) =========================
// One block owns 16 nodes. 8 waves; wave w computes gate cols [32w,32w+32).
// whh^T and wih^T fragments live in VGPRs for all 52 steps; h/c stay resident.
__global__ __launch_bounds__(256) void lstm_fused_kernel(
    const float* __restrict__ hseq,  // [T*N, 64] pre-ReLU GCN output
    const float* __restrict__ gcnb,  // [64]
    const float* __restrict__ wih,   // [256,64]
    const float* __restrict__ whh,   // [256,64]
    const float* __restrict__ bih,   // [256]
    const float* __restrict__ bhh,   // [256]
    const float* __restrict__ fcw,   // [64]
    const float* __restrict__ fcb,   // [1]
    float* __restrict__ out, int N) {
  __shared__ _Float16 hsm[16 * HID];      // recurrent h, f16 (WMMA A source)
  __shared__ _Float16 xsm[16 * HID];      // relu(hseq+b) tile, f16
  __shared__ float    gsm[16 * GATES];    // gate pre-activations
  __shared__ float    hfin[16 * HID];     // final h, f32 for FC

  const int tid   = threadIdx.x;
  const int lane  = tid & 31;
  const int wave  = tid >> 5;             // 0..7
  const int node0 = blockIdx.x * 16;
  const int nlo   = lane & 15;

  // Preload B fragments (weights transposed: B[k][col] = W[col*64+k]).
  v16h Bh[2][2], Bx[2][2];                // [col-tile][k-step]
  float biasc[2];
  #pragma unroll
  for (int ct = 0; ct < 2; ++ct) {
    int col = wave * 32 + ct * 16 + nlo;
    biasc[ct] = bih[col] + bhh[col];
    #pragma unroll
    for (int ks = 0; ks < 2; ++ks) {
      v16h bh, bx;
      #pragma unroll
      for (int j = 0; j < 8; ++j) {
        int k = ks * 32 + b_kidx(lane, j);
        bh[2*j]   = (_Float16)whh[(size_t)col * HID + k];
        bh[2*j+1] = (_Float16)whh[(size_t)col * HID + k + 1];
        bx[2*j]   = (_Float16)wih[(size_t)col * HID + k];
        bx[2*j+1] = (_Float16)wih[(size_t)col * HID + k + 1];
      }
      Bh[ct][ks] = bh;  Bx[ct][ks] = bx;
    }
  }

  // h0 = 0, c0 = 0 (c lives in registers: thread t owns elems t+256q).
  for (int i = tid; i < 16 * HID; i += 256) hsm[i] = (_Float16)0.f;
  float creg[4] = {0.f, 0.f, 0.f, 0.f};
  __syncthreads();

  for (int t = 0; t < TSTEPS; ++t) {
    // Stage ReLU(hseq + gcn_b) tile as f16 (node index clamped: garbage rows
    // are computed but never stored).
    #pragma unroll
    for (int q = 0; q < 4; ++q) {
      int i = tid + q * 256;
      int n = i >> 6, k = i & 63;
      int node = node0 + n;
      if (node >= N) node = N - 1;
      float v = hseq[((size_t)t * N + node) * HID + k] + gcnb[k];
      xsm[i] = (_Float16)(v > 0.f ? v : 0.f);
    }
    // Prefetch next timestep's tile while this step computes.
    if (t + 1 < TSTEPS) {
      const char* nxt = (const char*)(hseq + ((size_t)(t + 1) * N + node0) * HID);
      __builtin_prefetch(nxt + tid * 16, 0, 3);   // 256 thr x 16B = 4KB tile
    }
    __syncthreads();

    // Build A fragments from LDS (h and x tiles).
    v16h Ah[2], Ax[2];
    #pragma unroll
    for (int ks = 0; ks < 2; ++ks) {
      v16h ah, ax;
      #pragma unroll
      for (int j = 0; j < 8; ++j) {
        int k = ks * 32 + a_kidx(lane, j);
        ah[2*j] = hsm[nlo * HID + k];  ah[2*j+1] = hsm[nlo * HID + k + 1];
        ax[2*j] = xsm[nlo * HID + k];  ax[2*j+1] = xsm[nlo * HID + k + 1];
      }
      Ah[ks] = ah;  Ax[ks] = ax;
    }

    // gates = x@wih^T + h@whh^T + (bih+bhh) : 4 WMMA per 16x16 tile.
    #pragma unroll
    for (int ct = 0; ct < 2; ++ct) {
      v8f acc;
      #pragma unroll
      for (int r = 0; r < 8; ++r) acc[r] = biasc[ct];
      acc = __builtin_amdgcn_wmma_f32_16x16x32_f16(false, Ah[0], false, Bh[ct][0], (short)0, acc, false, false);
      acc = __builtin_amdgcn_wmma_f32_16x16x32_f16(false, Ah[1], false, Bh[ct][1], (short)0, acc, false, false);
      acc = __builtin_amdgcn_wmma_f32_16x16x32_f16(false, Ax[0], false, Bx[ct][0], (short)0, acc, false, false);
      acc = __builtin_amdgcn_wmma_f32_16x16x32_f16(false, Ax[1], false, Bx[ct][1], (short)0, acc, false, false);
      int mb  = (lane & 16) ? 8 : 0;
      int col = wave * 32 + ct * 16 + nlo;
      #pragma unroll
      for (int r = 0; r < 8; ++r) gsm[(mb + r) * GATES + col] = acc[r];
    }
    __syncthreads();

    // Cell update (PyTorch gate order i,f,g,o).
    #pragma unroll
    for (int q = 0; q < 4; ++q) {
      int i = tid + q * 256;
      int n = i >> 6, k = i & 63;
      float gi = gsm[n * GATES + k];
      float gf = gsm[n * GATES + 64 + k];
      float gg = gsm[n * GATES + 128 + k];
      float go = gsm[n * GATES + 192 + k];
      float c  = sigf(gf) * creg[q] + sigf(gi) * tanhf(gg);
      float h  = sigf(go) * tanhf(c);
      creg[q] = c;
      hsm[i] = (_Float16)h;
      if (t == TSTEPS - 1) hfin[i] = h;
    }
    __syncthreads();
  }

  // Final FC: out[n] = h_T @ fc_w + fc_b
  if (tid < 16) {
    int node = node0 + tid;
    if (node < N) {
      float acc = fcb[0];
      #pragma unroll
      for (int k = 0; k < HID; ++k) acc += hfin[tid * HID + k] * fcw[k];
      out[node] = acc;
    }
  }
}

// ================================ host ======================================
static void run_branch(const float* x, const int* edge, int N, int E,
                       const float* gw, const float* gb,
                       const float* wih, const float* whh,
                       const float* bih, const float* bhh,
                       const float* fcw, const float* fcb,
                       float* out, char* ws, size_t& off, hipStream_t stream) {
  size_t R = (size_t)N * TSTEPS;
  float* deg  = (float*)(ws + off); off += (((size_t)N * 4) + 255) & ~(size_t)255;
  float* xw   = (float*)(ws + off); off += ((R * HID * 4)   + 255) & ~(size_t)255;
  float* hseq = (float*)(ws + off); off += ((R * HID * 4)   + 255) & ~(size_t)255;

  const int* src = edge;
  const int* dst = edge + E;

  deg_init_kernel <<<(N + 255) / 256, 256, 0, stream>>>(deg, N);
  deg_accum_kernel<<<(E + 255) / 256, 256, 0, stream>>>(dst, deg, E);
  deg_rsqrt_kernel<<<(N + 255) / 256, 256, 0, stream>>>(deg, N);

  gemm_xw_kernel<<<(unsigned)((R + 15) / 16), 128, 0, stream>>>(x, gw, xw, (int)R);

  long long nh = (long long)N * HID;
  dim3 gInit((unsigned)((nh + 255) / 256), TSTEPS);
  hseq_init_kernel<<<gInit, 256, 0, stream>>>(xw, deg, hseq, N);

  dim3 gAgg((unsigned)((E + 3) / 4), TSTEPS);
  agg_edges_kernel<<<gAgg, 256, 0, stream>>>(src, dst, deg, xw, hseq, N, E);

  lstm_fused_kernel<<<(N + 15) / 16, 256, 0, stream>>>(hseq, gb, wih, whh, bih, bhh,
                                                       fcw, fcb, out, N);
}

extern "C" void kernel_launch(void* const* d_in, const int* in_sizes, int n_in,
                              void* d_out, int out_size, void* d_ws, size_t ws_size,
                              hipStream_t stream) {
  const float* xs[3]  = {(const float*)d_in[0], (const float*)d_in[2], (const float*)d_in[4]};
  const int*   es[3]  = {(const int*)d_in[1],   (const int*)d_in[3],   (const int*)d_in[5]};
  int Ns[3], Es[3];
  for (int b = 0; b < 3; ++b) {
    Ns[b] = in_sizes[2 * b]     / (TSTEPS * INCH);
    Es[b] = in_sizes[2 * b + 1] / 2;
  }
  const float* gcn_w = (const float*)d_in[6];    // [3,16,64]
  const float* gcn_b = (const float*)d_in[7];    // [3,64]
  const float* wih   = (const float*)d_in[8];    // [3,256,64]
  const float* whh   = (const float*)d_in[9];    // [3,256,64]
  const float* bih   = (const float*)d_in[10];   // [3,256]
  const float* bhh   = (const float*)d_in[11];   // [3,256]
  const float* fcw   = (const float*)d_in[12];   // [3,64,1]
  const float* fcb   = (const float*)d_in[13];   // [3,1]

  float* out = (float*)d_out;
  size_t off = 0;
  int outoff = 0;
  for (int b = 0; b < 3; ++b) {
    run_branch(xs[b], es[b], Ns[b], Es[b],
               gcn_w + (size_t)b * INCH * HID, gcn_b + (size_t)b * HID,
               wih + (size_t)b * GATES * HID,  whh + (size_t)b * GATES * HID,
               bih + (size_t)b * GATES,        bhh + (size_t)b * GATES,
               fcw + (size_t)b * HID,          fcb + b,
               out + outoff, (char*)d_ws, off, stream);
    outoff += Ns[b];
  }
  (void)n_in; (void)out_size; (void)ws_size;
}
